// LunaTransformerDecoderLayer_21088289423798
// MI455X (gfx1250) — compile-verified, matrix operand order checked by reading, not run on hardware
//
#include <hip/hip_runtime.h>
#include <hip/hip_bf16.h>

// ---------------------------------------------------------------------------
// Luna Transformer Decoder Layer for MI455X (gfx1250, wave32, WMMA)
// B=2 N=1024 PL=32 D=512 H=8 dh=64 F=2048
// GEMMs: bf16 WMMA, register-blocked 32x64 per wave, weight tile staged to
// LDS via CDNA5 async global->LDS loads (ASYNCcnt).
// ---------------------------------------------------------------------------

#define BB 2
#define NN 1024
#define PL 32
#define DD 512
#define HH 8
#define DH 64
#define FF 2048
#define M1 (BB*NN)
#define M2 (BB*PL)
#define LOG2C 0.6931471805599453f

// GEMM blocking
#define NB 64      // N columns per workgroup (4 N-tiles, shared via LDS)
#define MB 256     // M rows per workgroup (8 waves x 32 rows)
#define KC 64      // K chunk staged in LDS per iteration
#define LDP (KC + 8)   // row padding: 144B stride = 9 x 16B, conflict-free

typedef __attribute__((ext_vector_type(16))) __bf16 v16bf;
typedef __attribute__((ext_vector_type(8)))  float  v8f;
typedef __attribute__((ext_vector_type(4)))  unsigned int uint4v;

union FragBF { v16bf v; uint4v q[2]; };

__device__ __forceinline__ unsigned short f2bf(float f) {
    unsigned int u = __float_as_uint(f);
    unsigned int r = u + 0x7fffu + ((u >> 16) & 1u);   // round-to-nearest-even
    return (unsigned short)(r >> 16);
}

__device__ __forceinline__ float waveMax32(float v) {
    #pragma unroll
    for (int o = 16; o > 0; o >>= 1) v = fmaxf(v, __shfl_xor(v, o, 32));
    return v;
}
__device__ __forceinline__ float waveSum32(float v) {
    #pragma unroll
    for (int o = 16; o > 0; o >>= 1) v += __shfl_xor(v, o, 32);
    return v;
}

// ---------------- fp32 -> bf16 elementwise ----------------
__global__ void cvt_bf16_kernel(const float* __restrict__ in,
                                unsigned short* __restrict__ out, size_t n) {
    size_t i = (size_t)blockIdx.x * blockDim.x + threadIdx.x;
    if (i < n) out[i] = f2bf(in[i]);
}

// ---------------- transpose (K,N) fp32 -> (N,K) bf16 ----------------
__global__ void cvtT_bf16_kernel(const float* __restrict__ w,
                                 unsigned short* __restrict__ wt, int K, int N) {
    size_t i = (size_t)blockIdx.x * blockDim.x + threadIdx.x;
    size_t tot = (size_t)K * N;
    if (i >= tot) return;
    int n = (int)(i / K);
    int k = (int)(i - (size_t)n * K);
    wt[i] = f2bf(w[(size_t)k * N + n]);
}

// ---------------- bf16 WMMA GEMM -------------------------------------------
// C = alpha*(A*W^T + bias) [relu] [+resid]
// A: (M,K) bf16 row-major. Wt: (N,K) bf16 row-major.
// Workgroup: 256 threads = 8 waves, covers MB(256) x NB(64).
// Wave w: rows [w*32, w*32+32) x all 64 cols -> 2x4 = 8 WMMA tiles.
// Wt tile (64 x KC) staged into LDS with async global->LDS b128 loads.
__global__ void gemm_bf16_kernel(const unsigned short* __restrict__ A,
                                 const unsigned short* __restrict__ Wt,
                                 const float* __restrict__ bias,
                                 const float* __restrict__ resid,
                                 float* __restrict__ C,
                                 int M, int N, int K, float alpha, int relu) {
    __shared__ __align__(16) unsigned short bt[NB][LDP];
    int tid  = threadIdx.x;
    int wave = tid >> 5;
    int lane = tid & 31;
    int nBlocks = N / NB;
    int bm = blockIdx.x / nBlocks;
    int bn = blockIdx.x - bm * nBlocks;
    int nbase = bn * NB;
    int mbase = bm * MB + wave * 32;            // wave's 2 M-tiles
    if (mbase + 32 > M) mbase = (M > 32) ? (M - 32) : 0;   // clamp (small M)
    int half = lane >> 4;                        // K-half select per ISA layout
    int r16  = lane & 15;

    const unsigned short* arow0 = A + (size_t)(mbase + r16) * K;
    const unsigned short* arow1 = A + (size_t)(mbase + 16 + r16) * K;

    v8f zero = {};
    v8f acc[4][2];
    #pragma unroll
    for (int nt = 0; nt < 4; ++nt) { acc[nt][0] = zero; acc[nt][1] = zero; }

    for (int k0 = 0; k0 < K; k0 += KC) {
        // ---- async stage: Wt[nbase..+63][k0..k0+63] -> LDS (8KB, 2 x b128/thread)
        {
            int c = tid * 2;
            #pragma unroll
            for (int i = 0; i < 2; ++i) {
                int row = c >> 3;
                int cc  = (c & 7) * 8;
                const unsigned short* g = Wt + (size_t)(nbase + row) * K + k0 + cc;
                unsigned lds_off = (unsigned)(unsigned long long)&bt[row][cc];
                unsigned long long ga = (unsigned long long)g;
                asm volatile("global_load_async_to_lds_b128 %0, %1, off"
                             :: "v"(lds_off), "v"(ga) : "memory");
                ++c;
            }
        }
        asm volatile("s_wait_asynccnt 0x0" ::: "memory");
        __syncthreads();

        #pragma unroll
        for (int ks = 0; ks < KC; ks += 32) {
            int kb = k0 + ks + half * 8;
            FragBF a0, a1;
            a0.q[0] = *(const uint4v*)(arow0 + kb);
            a0.q[1] = *(const uint4v*)(arow0 + kb + 16);
            a1.q[0] = *(const uint4v*)(arow1 + kb);
            a1.q[1] = *(const uint4v*)(arow1 + kb + 16);
            int kbl = ks + half * 8;
            #pragma unroll
            for (int nt = 0; nt < 4; ++nt) {
                FragBF bf;
                const unsigned short* bp = &bt[nt * 16 + r16][kbl];
                bf.q[0] = *(const uint4v*)bp;
                bf.q[1] = *(const uint4v*)(bp + 16);
                acc[nt][0] = __builtin_amdgcn_wmma_f32_16x16x32_bf16(
                    false, a0.v, false, bf.v, (short)0, acc[nt][0], false, false);
                acc[nt][1] = __builtin_amdgcn_wmma_f32_16x16x32_bf16(
                    false, a1.v, false, bf.v, (short)0, acc[nt][1], false, false);
            }
        }
        __syncthreads();
    }

    // ---- epilogue ----
    #pragma unroll
    for (int nt = 0; nt < 4; ++nt) {
        int col = nbase + nt * 16 + r16;
        float bv = bias ? bias[col] : 0.f;
        #pragma unroll
        for (int t = 0; t < 2; ++t) {
            int rowbase = mbase + t * 16 + half * 8;
            #pragma unroll
            for (int r = 0; r < 8; ++r) {
                int row = rowbase + r;
                if (row < M) {
                    float v = alpha * (acc[nt][t][r] + bv);
                    if (relu) v = fmaxf(v, 0.f);
                    size_t idx = (size_t)row * N + col;
                    if (resid) v += resid[idx];
                    C[idx] = v;
                }
            }
        }
    }
}

// ---------------- pattn = softplus(ln2 * (pc @ pq^T)) / ln2 ----------------
__global__ void pattn_kernel(const float* __restrict__ pc,
                             const float* __restrict__ pq,
                             float* __restrict__ pattn) {
    int blk = blockIdx.x;
    int bh = blk >> 10, n = blk & (NN - 1);
    int b = bh >> 3, h = bh & 7;
    int j = threadIdx.x;
    __shared__ float pcr[DH];
    size_t base = ((size_t)(b * NN + n)) * DD + (size_t)h * DH;
    pcr[j] = pc[base + j];
    pcr[j + 32] = pc[base + j + 32];
    __syncthreads();
    const float* pqr = pq + ((size_t)(b * PL + j)) * DD + (size_t)h * DH;
    float s = 0.f;
    #pragma unroll
    for (int d = 0; d < DH; ++d) s = fmaf(pcr[d], pqr[d], s);
    float sp = LOG2C * s;
    float v = (sp > 15.f) ? sp : log1pf(__expf(sp));
    pattn[((size_t)bh * NN + n) * PL + j] = v / LOG2C;
}

// ---------------- fused ECA: aw=eca(q,k,pattn); mask; softmax; eca(probs,pattn,v)
__global__ void eca_fused_kernel(const float* __restrict__ q,
                                 const float* __restrict__ k,
                                 const float* __restrict__ v,
                                 const float* __restrict__ pattn,
                                 const unsigned char* __restrict__ pmask,
                                 float* __restrict__ attn) {
    int bh = blockIdx.x;
    int b = bh >> 3, h = bh & 7;
    int j = threadIdx.x;
    __shared__ float lk[DH], lq[DH], lp[PL], lprob[PL];
    float a1[DH], a2a[PL], a2b[PL];
    #pragma unroll
    for (int d = 0; d < DH; ++d) a1[d] = 0.f;
    #pragma unroll
    for (int d = 0; d < PL; ++d) { a2a[d] = 0.f; a2b[d] = 0.f; }
    bool masked = pmask[b * PL + j] != 0;
    size_t hoff = (size_t)h * DH;

    for (int n = 0; n < NN; ++n) {
        size_t base = ((size_t)(b * NN + n)) * DD + hoff;
        lk[j] = k[base + j]; lk[j + 32] = k[base + j + 32];
        lq[j] = q[base + j]; lq[j + 32] = q[base + j + 32];
        lp[j] = pattn[((size_t)bh * NN + n) * PL + j];
        float vj  = v[base + j];
        float vj2 = v[base + j + 32];
        __syncthreads();

        float pj = lp[j];
        float aw = 0.f;
        #pragma unroll
        for (int d = 0; d < DH; ++d) {
            a1[d] = fmaf(lk[d], pj, a1[d]);
            aw = fmaf(lq[d], a1[d], aw);
        }
        float invn = 1.f / (float)(n + 1);
        aw *= invn;
        if (masked) aw = 1e-9f;

        float m = waveMax32(aw);
        float e = __expf(aw - m);
        float ssum = waveSum32(e);
        lprob[j] = e / ssum;
        __syncthreads();

        float o1 = 0.f, o2 = 0.f;
        #pragma unroll
        for (int d1 = 0; d1 < PL; ++d1) {
            float pa = lp[d1];
            a2a[d1] = fmaf(pa, vj, a2a[d1]);
            a2b[d1] = fmaf(pa, vj2, a2b[d1]);
            float pr = lprob[d1];
            o1 = fmaf(pr, a2a[d1], o1);
            o2 = fmaf(pr, a2b[d1], o2);
        }
        attn[base + j] = o1 * invn;
        attn[base + j + 32] = o2 * invn;
        __syncthreads();
    }
}

// ---------------- MHA with Lq small (32), Lk=1024 (Yp = mha(p, enc, enc)) ---
__global__ void attn_qsmall_kernel(const float* __restrict__ q,
                                   const float* __restrict__ kp,
                                   const float* __restrict__ vp,
                                   const unsigned char* __restrict__ emask,
                                   float* __restrict__ ctx) {
    __shared__ float sc[NN];
    __shared__ float qrow[DH];
    __shared__ float red[128];
    int blk = blockIdx.x;
    int b = blk >> 8;
    int rem = blk & 255;
    int h = rem >> 5, qi = rem & 31;
    int tid = threadIdx.x;
    size_t qbase = ((size_t)(b * PL + qi)) * DD + (size_t)h * DH;
    if (tid < DH) qrow[tid] = q[qbase + tid];
    __syncthreads();

    for (int kk = tid; kk < NN; kk += 128) {
        const float* kr = kp + ((size_t)(b * NN + kk)) * DD + (size_t)h * DH;
        float s = 0.f;
        #pragma unroll
        for (int d = 0; d < DH; ++d) s = fmaf(qrow[d], kr[d], s);
        if (emask[b * NN + kk]) s = -10000.f;
        sc[kk] = s;
    }
    __syncthreads();

    float lm = -3.4e38f;
    for (int kk = tid; kk < NN; kk += 128) lm = fmaxf(lm, sc[kk]);
    red[tid] = lm; __syncthreads();
    for (int o = 64; o > 0; o >>= 1) { if (tid < o) red[tid] = fmaxf(red[tid], red[tid + o]); __syncthreads(); }
    float m = red[0];
    __syncthreads();

    float ls = 0.f;
    for (int kk = tid; kk < NN; kk += 128) { float e = __expf(sc[kk] - m); sc[kk] = e; ls += e; }
    red[tid] = ls; __syncthreads();
    for (int o = 64; o > 0; o >>= 1) { if (tid < o) red[tid] += red[tid + o]; __syncthreads(); }
    float sum = red[0];
    __syncthreads();

    if (tid < DH) {
        float acc = 0.f;
        for (int kk = 0; kk < NN; ++kk)
            acc = fmaf(sc[kk], vp[((size_t)(b * NN + kk)) * DD + (size_t)h * DH + tid], acc);
        ctx[qbase + tid] = acc / sum;
    }
}

// ---------------- MHA with Lk small (32), Lq=1024 (Yx = mha(dec, Yp, Yp)) ---
__global__ void attn_ksmall_kernel(const float* __restrict__ q,
                                   const float* __restrict__ kx,
                                   const float* __restrict__ vx,
                                   const unsigned char* __restrict__ pmask,
                                   float* __restrict__ ctx) {
    __shared__ float qrow[DH];
    __shared__ float pr[PL];
    int blk = blockIdx.x;
    int b = blk >> 13;
    int rem = blk & 8191;
    int h = rem >> 10, qi = rem & 1023;
    int j = threadIdx.x;
    size_t qbase = ((size_t)(b * NN + qi)) * DD + (size_t)h * DH;
    qrow[j] = q[qbase + j];
    qrow[j + 32] = q[qbase + j + 32];
    __syncthreads();

    const float* kr = kx + ((size_t)(b * PL + j)) * DD + (size_t)h * DH;
    float s = 0.f;
    #pragma unroll
    for (int d = 0; d < DH; ++d) s = fmaf(qrow[d], kr[d], s);
    if (pmask[b * PL + j]) s = -10000.f;
    float m = waveMax32(s);
    float e = __expf(s - m);
    float sum = waveSum32(e);
    pr[j] = e / sum;
    __syncthreads();

    float o1 = 0.f, o2 = 0.f;
    for (int jj = 0; jj < PL; ++jj) {
        const float* vr = vx + ((size_t)(b * PL + jj)) * DD + (size_t)h * DH;
        o1 = fmaf(pr[jj], vr[j], o1);
        o2 = fmaf(pr[jj], vr[j + 32], o2);
    }
    ctx[qbase + j] = o1;
    ctx[qbase + j + 32] = o2;
}

// ---------------- LayerNorm over D=512: out = LN(x + resid)*g + b -----------
__global__ void ln512_kernel(const float* __restrict__ x,
                             const float* __restrict__ resid,
                             const float* __restrict__ g,
                             const float* __restrict__ bb,
                             float* __restrict__ out) {
    __shared__ float xs[DD];
    __shared__ float red[128];
    int row = blockIdx.x, tid = threadIdx.x;
    const float* xr = x + (size_t)row * DD;
    const float* rr = resid + (size_t)row * DD;
    float s = 0.f;
    for (int i = tid; i < DD; i += 128) { float v = xr[i] + rr[i]; xs[i] = v; s += v; }
    red[tid] = s; __syncthreads();
    for (int o = 64; o > 0; o >>= 1) { if (tid < o) red[tid] += red[tid + o]; __syncthreads(); }
    float mean = red[0] * (1.f / DD);
    __syncthreads();
    float vs = 0.f;
    for (int i = tid; i < DD; i += 128) { float d = xs[i] - mean; vs += d * d; }
    red[tid] = vs; __syncthreads();
    for (int o = 64; o > 0; o >>= 1) { if (tid < o) red[tid] += red[tid + o]; __syncthreads(); }
    float inv = rsqrtf(red[0] * (1.f / DD) + 1e-5f);
    for (int i = tid; i < DD; i += 128)
        out[(size_t)row * DD + i] = (xs[i] - mean) * inv * g[i] + bb[i];
}

// ===========================================================================
extern "C" void kernel_launch(void* const* d_in, const int* in_sizes, int n_in,
                              void* d_out, int out_size, void* d_ws, size_t ws_size,
                              hipStream_t stream) {
    (void)in_sizes; (void)n_in; (void)out_size; (void)ws_size;
    const float* dec_in = (const float*)d_in[0];
    const float* p_in   = (const float*)d_in[1];
    const float* enc_in = (const float*)d_in[2];
    const unsigned char* p_mask   = (const unsigned char*)d_in[4];
    const unsigned char* enc_mask = (const unsigned char*)d_in[5];
    const float* pw[13]; const float* pbias[13];
    for (int i = 0; i < 13; ++i) { pw[i] = (const float*)d_in[6 + 2 * i]; pbias[i] = (const float*)d_in[7 + 2 * i]; }
    // 0 sa_q 1 sa_pq 2 sa_pc 3 sa_k 4 sa_v 5 sa_out 6 pk_q 7 pk_k 8 pk_v 9 up_q 10 up_k 11 up_v 12 ca_lin
    const float* yp_g = (const float*)d_in[32]; const float* yp_b = (const float*)d_in[33];
    const float* yx_g = (const float*)d_in[34]; const float* yx_b = (const float*)d_in[35];
    const float* fl_g = (const float*)d_in[36]; const float* fl_b = (const float*)d_in[37];
    const float* ff_w1 = (const float*)d_in[38]; const float* ff_b1 = (const float*)d_in[39];
    const float* ff_w2 = (const float*)d_in[40]; const float* ff_b2 = (const float*)d_in[41];
    float* out_f = (float*)d_out;
    float* yp_out = out_f + (size_t)M1 * DD;   // second tuple element

    // ---- workspace allocator ----
    char* base = (char*)d_ws;
    size_t off = 0;
    auto allocB = [&](size_t elems) -> unsigned short* {
        unsigned short* r = (unsigned short*)(base + off);
        off = (off + elems * 2 + 255) & ~(size_t)255; return r;
    };
    auto allocF = [&](size_t elems) -> float* {
        float* r = (float*)(base + off);
        off = (off + elems * 4 + 255) & ~(size_t)255; return r;
    };
    unsigned short* wt[13];
    for (int i = 0; i < 13; ++i) wt[i] = allocB((size_t)DD * DD);
    unsigned short* wtf1 = allocB((size_t)FF * DD);   // (2048,512)
    unsigned short* wtf2 = allocB((size_t)DD * FF);   // (512,2048)
    unsigned short* xb    = allocB((size_t)M1 * DD);
    unsigned short* pb    = allocB((size_t)M2 * DD);
    unsigned short* encb  = allocB((size_t)M1 * DD);
    unsigned short* attnb = allocB((size_t)M1 * DD);
    unsigned short* decb  = allocB((size_t)M1 * DD);
    unsigned short* Ypb   = allocB((size_t)M2 * DD);
    unsigned short* Yxcb  = allocB((size_t)M1 * DD);
    unsigned short* Yxnb  = allocB((size_t)M1 * DD);
    unsigned short* ff1b  = allocB((size_t)M1 * FF);
    float* qsa  = allocF((size_t)M1 * DD);
    float* ksa  = allocF((size_t)M1 * DD);
    float* vsa  = allocF((size_t)M1 * DD);
    float* pcf  = allocF((size_t)M1 * DD);
    float* pqf  = allocF((size_t)M2 * DD);
    float* patt = allocF((size_t)BB * HH * NN * PL);
    float* attn = allocF((size_t)M1 * DD);
    float* dec  = allocF((size_t)M1 * DD);
    float* qp   = allocF((size_t)M2 * DD);
    float* kp   = allocF((size_t)M1 * DD);
    float* vp   = allocF((size_t)M1 * DD);
    float* Ypc  = allocF((size_t)M2 * DD);
    float* qx   = allocF((size_t)M1 * DD);
    float* kx   = allocF((size_t)M2 * DD);
    float* vx   = allocF((size_t)M2 * DD);
    float* Yxc  = allocF((size_t)M1 * DD);
    float* Yx   = allocF((size_t)M1 * DD);
    float* Yxn  = allocF((size_t)M1 * DD);
    float* ff1  = allocF((size_t)M1 * FF);
    float* ff2  = allocF((size_t)M1 * DD);

    auto cvt = [&](const float* s, unsigned short* d, size_t n) {
        cvt_bf16_kernel<<<(unsigned)((n + 255) / 256), 256, 0, stream>>>(s, d, n);
    };
    auto cvtT = [&](const float* w, unsigned short* d, int K, int N) {
        size_t tot = (size_t)K * N;
        cvtT_bf16_kernel<<<(unsigned)((tot + 255) / 256), 256, 0, stream>>>(w, d, K, N);
    };
    auto gemm = [&](const unsigned short* A, const unsigned short* Wt, const float* bias,
                    const float* resid, float* C, int M, int N, int K, float alpha, int relu) {
        int mBlocks = (M + MB - 1) / MB;
        int nBlocks = N / NB;
        gemm_bf16_kernel<<<mBlocks * nBlocks, 256, 0, stream>>>(A, Wt, bias, resid, C, M, N, K, alpha, relu);
    };
    const float scale = 0.125f;   // dh^-0.5, dh = 64

    // ---- weight + activation conversion ----
    for (int i = 0; i < 13; ++i) cvtT(pw[i], wt[i], DD, DD);
    cvtT(ff_w1, wtf1, DD, FF);
    cvtT(ff_w2, wtf2, FF, DD);
    cvt(dec_in, xb, (size_t)M1 * DD);
    cvt(p_in, pb, (size_t)M2 * DD);
    cvt(enc_in, encb, (size_t)M1 * DD);

    // ---- Luna causal self-attention ----
    gemm(xb, wt[0], pbias[0], nullptr, qsa, M1, DD, DD, scale, 0);  // q (scaled)
    gemm(pb, wt[1], pbias[1], nullptr, pqf, M2, DD, DD, scale, 0);  // pq (scaled)
    gemm(xb, wt[2], pbias[2], nullptr, pcf, M1, DD, DD, 1.f, 0);    // pc
    gemm(xb, wt[3], pbias[3], nullptr, ksa, M1, DD, DD, 1.f, 0);    // k
    gemm(xb, wt[4], pbias[4], nullptr, vsa, M1, DD, DD, 1.f, 0);    // v
    pattn_kernel<<<BB * HH * NN, 32, 0, stream>>>(pcf, pqf, patt);
    eca_fused_kernel<<<BB * HH, 32, 0, stream>>>(qsa, ksa, vsa, patt, p_mask, attn);
    cvt(attn, attnb, (size_t)M1 * DD);
    gemm(attnb, wt[5], pbias[5], dec_in, dec, M1, DD, DD, 1.f, 0);  // out proj + residual
    cvt(dec, decb, (size_t)M1 * DD);

    // ---- Yp = mha(p, enc, enc) ----
    gemm(pb,   wt[6], pbias[6], nullptr, qp, M2, DD, DD, scale, 0);
    gemm(encb, wt[7], pbias[7], nullptr, kp, M1, DD, DD, 1.f, 0);
    gemm(encb, wt[8], pbias[8], nullptr, vp, M1, DD, DD, 1.f, 0);
    attn_qsmall_kernel<<<BB * HH * PL, 128, 0, stream>>>(qp, kp, vp, enc_mask, Ypc);
    cvt(Ypc, Ypb, (size_t)M2 * DD);

    // ---- Yx = mha(dec, Yp, Yp), then ca_lin ----
    gemm(decb, wt[9],  pbias[9],  nullptr, qx, M1, DD, DD, scale, 0);
    gemm(Ypb,  wt[10], pbias[10], nullptr, kx, M2, DD, DD, 1.f, 0);
    gemm(Ypb,  wt[11], pbias[11], nullptr, vx, M2, DD, DD, 1.f, 0);
    attn_ksmall_kernel<<<BB * HH * NN, 32, 0, stream>>>(qx, kx, vx, p_mask, Yxc);
    cvt(Yxc, Yxcb, (size_t)M1 * DD);
    gemm(Yxcb, wt[12], pbias[12], nullptr, Yx, M1, DD, DD, 1.f, 0);

    // ---- LayerNorms (Yp output goes straight to d_out) ----
    ln512_kernel<<<M2, 128, 0, stream>>>(Ypc, p_in, yp_g, yp_b, yp_out);
    ln512_kernel<<<M1, 128, 0, stream>>>(Yx, dec, yx_g, yx_b, Yxn);

    // ---- FFN + final LN ----
    cvt(Yxn, Yxnb, (size_t)M1 * DD);
    gemm(Yxnb, wtf1, ff_b1, nullptr, ff1, M1, FF, DD, 1.f, 1);   // relu
    cvt(ff1, ff1b, (size_t)M1 * FF);
    gemm(ff1b, wtf2, ff_b2, nullptr, ff2, M1, DD, FF, 1.f, 0);
    ln512_kernel<<<M1, 128, 0, stream>>>(ff2, Yxn, fl_g, fl_b, out_f);
}
